// AttnBlock_180388627347
// MI455X (gfx1250) — compile-verified, matrix operand order checked by reading, not run on hardware
//
#include <hip/hip_runtime.h>
#include <hip/hip_bf16.h>

typedef __attribute__((ext_vector_type(16))) _Float16 v16h;
typedef __attribute__((ext_vector_type(8)))  _Float16 v8h;
typedef __attribute__((ext_vector_type(8)))  float    v8f;
typedef __attribute__((ext_vector_type(4)))  float    v4f;

#define EMB   256
#define HEADS 4
#define HD    64
#define BATCH 4
#define SEQ   4096
#define TOK   (BATCH*SEQ)

static_assert(sizeof(_Float16) == 2, "f16 size");

// ---------------------------------------------------------------------------
// Cross-lane XOR exchange without DS traffic: dpp8 for masks 1/2/4,
// v_permlane16 for mask 8, v_permlanex16 for mask 16.  Pure VALU.
// ---------------------------------------------------------------------------
constexpr int dpp8_sel(int m) {
  int s = 0;
  for (int i = 0; i < 8; ++i) s |= ((i ^ m) & 7) << (3 * i);
  return s;
}

template<int M>
__device__ __forceinline__ float lane_xor(float v) {
  int x = __builtin_bit_cast(int, v);
  int y;
  if constexpr (M < 8) {
    y = __builtin_amdgcn_mov_dpp8(x, dpp8_sel(M));
  } else if constexpr (M == 8) {
    y = __builtin_amdgcn_permlane16(x, x, (int)0xFEDCBA98u, (int)0x76543210u,
                                    false, false);
  } else {  // 16: swap 16-lane rows
    y = __builtin_amdgcn_permlanex16(x, x, (int)0x76543210u, (int)0xFEDCBA98u,
                                     false, false);
  }
  return __builtin_bit_cast(float, y);
}

__device__ __forceinline__ float red_max16(float v) {
  v = fmaxf(v, lane_xor<1>(v));
  v = fmaxf(v, lane_xor<2>(v));
  v = fmaxf(v, lane_xor<4>(v));
  v = fmaxf(v, lane_xor<8>(v));
  return v;
}
__device__ __forceinline__ float red_sum16(float v) {
  v += lane_xor<1>(v); v += lane_xor<2>(v);
  v += lane_xor<4>(v); v += lane_xor<8>(v);
  return v;
}
__device__ __forceinline__ float red_sum32(float v) {
  v = red_sum16(v);
  v += lane_xor<16>(v);
  return v;
}

// ---------------------------------------------------------------------------
// Async global->LDS copy (CDNA5, ASYNCcnt-tracked, no VGPR round trip).
// Flat aperture maps LDS as addr[31:0] (ISA 10.2), so truncating the generic
// pointer to a __shared__ object yields the hardware LDS byte address.
// ---------------------------------------------------------------------------
__device__ __forceinline__ void async_ld_b128(const void* gptr, void* lptr) {
  uint32_t l = (uint32_t)(uintptr_t)lptr;
  asm volatile("global_load_async_to_lds_b128 %0, %1, off"
               :: "v"(l), "v"(gptr)
               : "memory");
}
__device__ __forceinline__ void wait_async0() {
  asm volatile("s_wait_asynccnt 0x0" ::: "memory");
}

// ---------------------------------------------------------------------------
// WMMA operand loader (A-operand; identical addressing for B when the source
// is stored with the N dimension as rows).
// ---------------------------------------------------------------------------
__device__ __forceinline__ v16h load_op(const _Float16* p, int half) {
  v8h lo = *(const v8h*)(p + half * 8);
  v8h hi = *(const v8h*)(p + half * 8 + 16);
  return __builtin_shufflevector(lo, hi, 0,1,2,3,4,5,6,7,8,9,10,11,12,13,14,15);
}

__device__ __forceinline__ v8f wmma32(v16h a, v16h b, v8f c) {
  return __builtin_amdgcn_wmma_f32_16x16x32_f16(false, a, false, b, (short)0, c,
                                                false, false);
}

// ---------------------------------------------------------------------------
// LayerNorm: one wave per token, 8 contiguous channels per lane.
// MODE 0: input x[b][c][s] (strided gather); MODE 1: input av[t][c].
// ---------------------------------------------------------------------------
template<int MODE>
__global__ __launch_bounds__(256) void ln_k(const float* __restrict__ in,
                                            const float* __restrict__ gam,
                                            const float* __restrict__ bet,
                                            _Float16* __restrict__ out) {
  int t    = blockIdx.x * 8 + (threadIdx.x >> 5);
  int lane = threadIdx.x & 31;
  float v[8];
  if (MODE == 0) {
    int b = t >> 12, s = t & 4095;
    const float* p = in + ((size_t)b * EMB + lane * 8) * SEQ + s;
#pragma unroll
    for (int r = 0; r < 8; ++r) v[r] = p[(size_t)r * SEQ];
  } else {
    const float* p = in + (size_t)t * EMB + lane * 8;
#pragma unroll
    for (int r = 0; r < 8; ++r) v[r] = p[r];
  }
  float sum = 0.f, sq = 0.f;
#pragma unroll
  for (int r = 0; r < 8; ++r) { sum += v[r]; sq += v[r] * v[r]; }
  sum = red_sum32(sum);
  sq  = red_sum32(sq);
  float mu  = sum * (1.f / EMB);
  float var = sq * (1.f / EMB) - mu * mu;
  float rs  = rsqrtf(var + 1e-5f);
  _Float16 o[8];
#pragma unroll
  for (int r = 0; r < 8; ++r) {
    int c = lane * 8 + r;
    o[r] = (_Float16)((v[r] - mu) * rs * gam[c] + bet[c]);
  }
  *(v8h*)(out + (size_t)t * EMB + lane * 8) = *(const v8h*)o;
}

// ---------------------------------------------------------------------------
// Generic WMMA GEMM, block = 8 waves, tile M=128 N=64, grid (SEQ/128,4,BATCH).
// EPI: 0 = f16 [b,h,s,64] (Q/K)       1 = f16 [b,h,64,s] (V transposed)
//      2 = f32 av = val+bias+x^T       3 = f16 gelu(val+bias)
//      4 = f32 out[b][c][s] = val+bias+av
// ---------------------------------------------------------------------------
template<int EPI, int KLEN>
__global__ __launch_bounds__(256) void gemm_k(
    const _Float16* __restrict__ A, int lda, int kst_per_nb,
    const float* __restrict__ W, int ldw, int wcol_per_nb,
    const float* __restrict__ bias, int bias_per_nb,
    _Float16* __restrict__ C16, float* __restrict__ C32,
    const float* __restrict__ aux) {
  __shared__ __align__(16) _Float16 wt[64][KLEN + 8];  // wt[n][k], padded
  const int b = blockIdx.z, nb = blockIdx.y;
  const int tid = threadIdx.x, lane = tid & 31, w = tid >> 5;
  const int half = lane >> 4, l16 = lane & 15;
  const int k0 = kst_per_nb * nb, col0 = wcol_per_nb * nb;

  for (int e = tid; e < 64 * KLEN; e += 256) {   // coalesced over n
    int n = e & 63, k = e >> 6;
    wt[n][k] = (_Float16)W[(size_t)k * ldw + col0 + n];
  }
  __syncthreads();

  const _Float16* Ab = A + (size_t)b * SEQ * lda;
  const int m0  = blockIdx.x * 128 + w * 16;
  const int row = m0 + l16;

  v8f acc[4] = {};
#pragma unroll 2
  for (int kt = 0; kt < KLEN / 32; ++kt) {
    v16h a = load_op(Ab + (size_t)row * lda + k0 + kt * 32, half);
#pragma unroll
    for (int nt = 0; nt < 4; ++nt) {
      v16h bb = load_op(&wt[nt * 16 + l16][kt * 32], half);
      acc[nt] = wmma32(a, bb, acc[nt]);
    }
  }

  const float* bp = bias + bias_per_nb * nb;
  const int mb = m0 + half * 8;
#pragma unroll
  for (int nt = 0; nt < 4; ++nt) {
    int nloc = nt * 16 + l16;
    v8f vv = acc[nt] + bp[nloc];
    if (EPI == 0) {
      _Float16* dst = C16 + ((size_t)(b * HEADS + nb) * SEQ + mb) * HD + nloc;
#pragma unroll
      for (int r = 0; r < 8; ++r) dst[(size_t)r * HD] = (_Float16)vv[r];
    } else if (EPI == 1) {
      _Float16 tmp[8];
#pragma unroll
      for (int r = 0; r < 8; ++r) tmp[r] = (_Float16)vv[r];
      *(v8h*)(C16 + ((size_t)(b * HEADS + nb) * HD + nloc) * SEQ + mb) =
          *(const v8h*)tmp;
    } else if (EPI == 2) {
      int col = nb * 64 + nloc;
      const float* xp = aux + ((size_t)b * EMB + col) * SEQ + mb;
      v4f x0 = *(const v4f*)xp, x1 = *(const v4f*)(xp + 4);
      float* dst = C32 + ((size_t)b * SEQ + mb) * EMB + col;
#pragma unroll
      for (int r = 0; r < 4; ++r) dst[(size_t)r * EMB] = vv[r] + x0[r];
#pragma unroll
      for (int r = 0; r < 4; ++r) dst[(size_t)(r + 4) * EMB] = vv[r + 4] + x1[r];
    } else if (EPI == 3) {
      int col = nb * 64 + nloc;
      _Float16* dst = C16 + ((size_t)b * SEQ + mb) * EMB + col;
#pragma unroll
      for (int r = 0; r < 8; ++r) {
        float g = 0.5f * vv[r] * (1.0f + erff(vv[r] * 0.70710678118f));
        dst[(size_t)r * EMB] = (_Float16)g;
      }
    } else {
      int col = nb * 64 + nloc;
      const float* ap = aux + ((size_t)b * SEQ + mb) * EMB + col;
      v8f s = vv;
#pragma unroll
      for (int r = 0; r < 8; ++r) s[r] += ap[(size_t)r * EMB];
      float* dst = C32 + ((size_t)b * EMB + col) * SEQ + mb;
      *(v4f*)dst       = __builtin_shufflevector(s, s, 0, 1, 2, 3);
      *(v4f*)(dst + 4) = __builtin_shufflevector(s, s, 4, 5, 6, 7);
    }
  }
}

// ---------------------------------------------------------------------------
// Flash attention, grid (SEQ/128, HEADS, BATCH), 8 waves/block.
// K/V^T staged with global_load_async_to_lds_b128 into the alternate LDS
// buffer while the current one is consumed; s_wait_asynccnt + barrier per
// iteration.  Softmax reductions are pure-VALU (dpp8/permlane16).
// ---------------------------------------------------------------------------
__global__ __launch_bounds__(256) void attn_k(const _Float16* __restrict__ q,
                                              const _Float16* __restrict__ k,
                                              const _Float16* __restrict__ vt,
                                              _Float16* __restrict__ o) {
  __shared__ __align__(16) _Float16 kbuf[2][64][72];
  __shared__ __align__(16) _Float16 vbuf[2][64][72];
  __shared__ __align__(16) _Float16 pbuf[8][16][72];
  const int b = blockIdx.z, h = blockIdx.y;
  const int tid = threadIdx.x, lane = tid & 31, w = tid >> 5;
  const int half = lane >> 4, l16 = lane & 15;
  const size_t hb = (size_t)b * HEADS + h;
  const _Float16* qh = q  + hb * SEQ * HD;   // [s][d]
  const _Float16* kh = k  + hb * SEQ * HD;   // [s][d]
  const _Float16* vh = vt + hb * HD * SEQ;   // [d][s]
  const int m0 = blockIdx.x * 128 + w * 16;
  const int r0 = tid >> 3, o0 = (tid & 7) * 8;   // staging chunk coords
  const int NBLK = SEQ / 64;

  v16h qa0 = load_op(qh + (size_t)(m0 + l16) * HD + 0,  half);
  v16h qa1 = load_op(qh + (size_t)(m0 + l16) * HD + 32, half);

  v8f oacc[4] = {};
  float mi[8], li[8];
#pragma unroll
  for (int r = 0; r < 8; ++r) { mi[r] = -1e30f; li[r] = 0.f; }
  const float scale = 0.125f;  // 1/sqrt(64)

  auto stage = [&](int jj, int bufi) {
    const _Float16* kp = kh + (size_t)(jj * 64 + r0) * HD + o0;
    const _Float16* vp = vh + (size_t)r0 * SEQ + jj * 64 + o0;
    async_ld_b128(kp,                    &kbuf[bufi][r0][o0]);
    async_ld_b128(kp + (size_t)32 * HD,  &kbuf[bufi][r0 + 32][o0]);
    async_ld_b128(vp,                    &vbuf[bufi][r0][o0]);
    async_ld_b128(vp + (size_t)32 * SEQ, &vbuf[bufi][r0 + 32][o0]);
  };

  stage(0, 0);
  wait_async0();
  __syncthreads();

  for (int j = 0; j < NBLK; ++j) {
    // prefetch two blocks ahead (global_prefetch_b8)
    if (j + 2 < NBLK) {
      if (tid < 64)
        __builtin_prefetch(kh + (size_t)((j + 2) * 64 + tid) * HD, 0, 1);
      else if (tid < 128)
        __builtin_prefetch(vh + (size_t)(tid - 64) * SEQ + (j + 2) * 64, 0, 1);
    }
    // async-stage next block into the alternate buffer (safe: last read of it
    // was fenced by the previous iteration's barrier)
    if (j + 1 < NBLK) stage(j + 1, (j + 1) & 1);

    const _Float16 (*kcur)[72] = kbuf[j & 1];
    const _Float16 (*vcur)[72] = vbuf[j & 1];

    // S = Q @ K^T; split per K-chunk so accumulator reuse distance is 4 WMMAs
    v8f sc[4];
    v8f zero = {};
#pragma unroll
    for (int nt = 0; nt < 4; ++nt) {
      v16h b0 = load_op(&kcur[nt * 16 + l16][0], half);
      sc[nt] = wmma32(qa0, b0, zero);
    }
#pragma unroll
    for (int nt = 0; nt < 4; ++nt) {
      v16h b1 = load_op(&kcur[nt * 16 + l16][32], half);
      sc[nt] = wmma32(qa1, b1, sc[nt]);
    }

    // online softmax
    float fr[8], mnew[8];
#pragma unroll
    for (int r = 0; r < 8; ++r) {
      float mx = fmaxf(fmaxf(sc[0][r], sc[1][r]), fmaxf(sc[2][r], sc[3][r]));
      mx = red_max16(mx);
      float mn = fmaxf(mi[r], mx * scale);
      fr[r] = __expf(mi[r] - mn);
      mnew[r] = mn;
      mi[r] = mn;
    }
    float rs[8];
#pragma unroll
    for (int r = 0; r < 8; ++r) rs[r] = 0.f;
#pragma unroll
    for (int nt = 0; nt < 4; ++nt) {
#pragma unroll
      for (int r = 0; r < 8; ++r) {
        float p = __expf(sc[nt][r] * scale - mnew[r]);
        rs[r] += p;
        pbuf[w][r + half * 8][nt * 16 + l16] = (_Float16)p;
      }
    }
#pragma unroll
    for (int r = 0; r < 8; ++r) li[r] = li[r] * fr[r] + red_sum16(rs[r]);
#pragma unroll
    for (int nt = 0; nt < 4; ++nt)
#pragma unroll
      for (int r = 0; r < 8; ++r) oacc[nt][r] *= fr[r];

    // O += P @ V^T (per-wave pbuf: same-wave LDS ordering, no barrier)
    v16h pa0 = load_op(&pbuf[w][l16][0],  half);
    v16h pa1 = load_op(&pbuf[w][l16][32], half);
#pragma unroll
    for (int nt = 0; nt < 4; ++nt) {
      v16h b0 = load_op(&vcur[nt * 16 + l16][0], half);
      oacc[nt] = wmma32(pa0, b0, oacc[nt]);
    }
#pragma unroll
    for (int nt = 0; nt < 4; ++nt) {
      v16h b1 = load_op(&vcur[nt * 16 + l16][32], half);
      oacc[nt] = wmma32(pa1, b1, oacc[nt]);
    }

    if (j + 1 < NBLK) {
      wait_async0();      // this wave's async copies into the alt buffer done
      __syncthreads();    // all waves' copies visible before next iteration
    }
  }

  // normalize and store merged-head layout [b][s][EMB] as f16
#pragma unroll
  for (int r = 0; r < 8; ++r) {
    float inv = 1.0f / li[r];
    int m = m0 + r + half * 8;
#pragma unroll
    for (int nt = 0; nt < 4; ++nt) {
      o[((size_t)b * SEQ + m) * EMB + h * 64 + nt * 16 + l16] =
          (_Float16)(oacc[nt][r] * inv);
    }
  }
}

// ---------------------------------------------------------------------------
extern "C" void kernel_launch(void* const* d_in, const int* in_sizes, int n_in,
                              void* d_out, int out_size, void* d_ws,
                              size_t ws_size, hipStream_t stream) {
  const float* x    = (const float*)d_in[0];
  const float* ln1g = (const float*)d_in[1];
  const float* ln1b = (const float*)d_in[2];
  const float* wq   = (const float*)d_in[3];
  const float* bq   = (const float*)d_in[4];
  const float* wk   = (const float*)d_in[5];
  const float* bk   = (const float*)d_in[6];
  const float* wv   = (const float*)d_in[7];
  const float* bv   = (const float*)d_in[8];
  const float* wo   = (const float*)d_in[9];
  const float* bo   = (const float*)d_in[10];
  const float* ln2g = (const float*)d_in[11];
  const float* ln2b = (const float*)d_in[12];
  const float* w1   = (const float*)d_in[13];
  const float* b1   = (const float*)d_in[14];
  const float* w2   = (const float*)d_in[15];
  const float* b2   = (const float*)d_in[16];
  float* out = (float*)d_out;

  char* ws = (char*)d_ws;
  const size_t F16SZ = (size_t)TOK * EMB * sizeof(_Float16);  // 8 MB
  _Float16* xn  = (_Float16*)(ws + 0 * F16SZ);  // LN1 out; reused as LN2 out
  _Float16* qb  = (_Float16*)(ws + 1 * F16SZ);
  _Float16* kb  = (_Float16*)(ws + 2 * F16SZ);
  _Float16* vtb = (_Float16*)(ws + 3 * F16SZ);
  _Float16* ob  = (_Float16*)(ws + 4 * F16SZ);  // attn out; reused as ff1
  float*    av  = (float*)   (ws + 5 * F16SZ);  // 16 MB

  dim3 blk(256);
  dim3 gln(TOK / 8);
  dim3 gg(SEQ / 128, 4, BATCH);

  ln_k<0><<<gln, blk, 0, stream>>>(x, ln1g, ln1b, xn);

  gemm_k<0, 64><<<gg, blk, 0, stream>>>(xn, EMB, 64, wq, HD, 0, bq, 0,
                                        qb, nullptr, nullptr);
  gemm_k<0, 64><<<gg, blk, 0, stream>>>(xn, EMB, 64, wk, HD, 0, bk, 0,
                                        kb, nullptr, nullptr);
  gemm_k<1, 64><<<gg, blk, 0, stream>>>(xn, EMB, 64, wv, HD, 0, bv, 0,
                                        vtb, nullptr, nullptr);

  attn_k<<<gg, blk, 0, stream>>>(qb, kb, vtb, ob);

  gemm_k<2, 256><<<gg, blk, 0, stream>>>(ob, EMB, 0, wo, EMB, 64, bo, 64,
                                         nullptr, av, x);

  ln_k<1><<<gln, blk, 0, stream>>>(av, ln2g, ln2b, xn);

  gemm_k<3, 256><<<gg, blk, 0, stream>>>(xn, EMB, 0, w1, EMB, 64, b1, 64,
                                         ob, nullptr, nullptr);
  gemm_k<4, 256><<<gg, blk, 0, stream>>>(ob, EMB, 0, w2, EMB, 64, b2, 64,
                                         nullptr, out, av);
}
